// VACNN_17205638988182
// MI455X (gfx1250) — matrix operand order repaired and checked by reading, not compiled, not run on hardware
//
#include <hip/hip_runtime.h>
#include <hip/hip_bf16.h>
#include <cstddef>

// ---------------------------------------------------------------------------
// MI455X (gfx1250, wave32). All activations NHWC f16, weights O x (KH*KW*C)
// f16, GEMM via v_wmma_f32_16x16x32_f16 with fused BN/residual/ReLU epilogue.
// 128x128x32 block tile, 8 waves (2x4), each wave 4x2 WMMA tiles.
// Tile fill uses GLOBAL_LOAD_ASYNC_TO_LDS_B128 (ASYNCcnt) — builtin confirmed
// present; signature takes v4i addrspace(1)* / addrspace(3)* pointers.
// LDS rows padded to 40 halfs to spread the 64 banks.
// ---------------------------------------------------------------------------

typedef __attribute__((ext_vector_type(16))) _Float16 v16h;
typedef __attribute__((ext_vector_type(8)))  _Float16 v8h;
typedef __attribute__((ext_vector_type(8)))  float    v8f;
typedef __attribute__((ext_vector_type(4)))  int      v4i;

#define BM 128
#define BN 128
#define BK 32
#define BKP (BK + 8)   // +16B row pad: bank stride 20 on 64 banks

#if __has_builtin(__builtin_amdgcn_global_load_async_to_lds_b128)
#define HAVE_ASYNC_LDS 1
#define ASYNC_B128(gp, lp)                                                    \
  __builtin_amdgcn_global_load_async_to_lds_b128(                             \
      (__attribute__((address_space(1))) v4i*)(gp),                           \
      (__attribute__((address_space(3))) v4i*)(lp), 0, 0)
#endif

__device__ __forceinline__ void wait_async_lds() {
#ifdef HAVE_ASYNC_LDS
#if __has_builtin(__builtin_amdgcn_s_wait_asynccnt)
  __builtin_amdgcn_s_wait_asynccnt(0);
#else
  asm volatile("s_wait_asynccnt 0x0" ::: "memory");
#endif
#endif
}

__device__ __forceinline__ float h2f(_Float16 h) { return (float)h; }

// ------------------------- implicit GEMM conv (WMMA) -----------------------
// act NHWC: [N][H][W][C]; wgt: [O][KH*KW*C] (k = (ky*KW+kx)*C + c)
// out NHWC: [N][OH][OW][O];  C[o][p] += W[o][k] * im2col[k][p]
__global__ __launch_bounds__(256) void k_conv_wmma(
    const _Float16* __restrict__ wgt,
    const _Float16* __restrict__ act,
    const float* __restrict__ bn_scale,
    const float* __restrict__ bn_bias,
    const _Float16* __restrict__ resid,   // NHWC, same shape as out, or null
    _Float16* __restrict__ out_h,
    float* __restrict__ out_f,
    int Nb, int C, int H, int W,
    int O, int KH, int KW,
    int stride, int pad,
    int OH, int OW, int relu)
{
  const int Ktot = C * KH * KW;
  const int P    = Nb * OH * OW;

  __shared__ __align__(32) _Float16 As[BM][BKP];  // weights (M-major)
  __shared__ __align__(32) _Float16 Bs[BN][BKP];  // im2col, transposed (N-major)

  const int tid  = threadIdx.x;
  const int wave = tid >> 5;
  const int lane = tid & 31;
  const int wm   = wave >> 2;
  const int wn   = wave & 3;
  const int m0   = blockIdx.y * BM;
  const int p0   = blockIdx.x * BN;

  v8f acc[4][2] = {};

  const int r  = tid >> 1;
  const int kk = (tid & 1) * 16;

  // hoisted output-pixel decomposition for this thread's B row
  const int pB   = p0 + r;
  const bool pok = (pB < P);
  int nimg = 0, oy = 0, ox = 0;
  if (pok) {
    nimg = pB / (OH * OW);
    int pr = pB - nimg * (OH * OW);
    oy = pr / OW;
    ox = pr - oy * OW;
  }
  const int iy0 = oy * stride - pad;
  const int ix0 = ox * stride - pad;
  const int oA  = m0 + r;
  const bool vecA = ((Ktot & 7) == 0);
  const bool vecB = ((C & 15) == 0);     // 16-k window stays inside one C-run

  for (int k0 = 0; k0 < Ktot; k0 += BK) {
    // ---- A tile (weights, row-major) ----
    if (vecA && oA < O && k0 + kk + 16 <= Ktot) {
      const _Float16* g = wgt + (size_t)oA * Ktot + (k0 + kk);
#ifdef HAVE_ASYNC_LDS
      ASYNC_B128(g,     &As[r][kk]);
      ASYNC_B128(g + 8, &As[r][kk + 8]);
#else
      *(v8h*)&As[r][kk]     = *(const v8h*)g;
      *(v8h*)&As[r][kk + 8] = *(const v8h*)(g + 8);
#endif
    } else {
#pragma unroll
      for (int t2 = 0; t2 < 2; ++t2) {
        int kg = k0 + kk + t2 * 8;
        v8h av = {0, 0, 0, 0, 0, 0, 0, 0};
        if (oA < O) {
#pragma unroll
          for (int e = 0; e < 8; ++e)
            if (kg + e < Ktot) av[e] = wgt[(size_t)oA * Ktot + kg + e];
        }
        *(v8h*)&As[r][kk + t2 * 8] = av;
      }
    }

    // ---- B tile (implicit im2col from NHWC) ----
    {
      int kg0 = k0 + kk;
      if (vecB && pok && kg0 + 16 <= Ktot) {
        int q  = kg0 / C;              // spatial kernel index (ky,kx)
        int c0 = kg0 - q * C;
        int ky = q / KW;
        int kx = q - ky * KW;
        int iy = iy0 + ky, ix = ix0 + kx;
        if ((unsigned)iy < (unsigned)H && (unsigned)ix < (unsigned)W) {
          const _Float16* src = act + ((size_t)(nimg * H + iy) * W + ix) * C + c0;
#ifdef HAVE_ASYNC_LDS
          ASYNC_B128(src,     &Bs[r][kk]);
          ASYNC_B128(src + 8, &Bs[r][kk + 8]);
#else
          *(v8h*)&Bs[r][kk]     = *(const v8h*)src;
          *(v8h*)&Bs[r][kk + 8] = *(const v8h*)(src + 8);
#endif
        } else {
          v8h z = {0, 0, 0, 0, 0, 0, 0, 0};
          *(v8h*)&Bs[r][kk]     = z;
          *(v8h*)&Bs[r][kk + 8] = z;
        }
      } else {
#pragma unroll 4
        for (int e = 0; e < 16; ++e) {
          int kg = kg0 + e;
          _Float16 v = (_Float16)0.0f;
          if (pok && kg < Ktot) {
            int q  = kg / C;
            int c  = kg - q * C;
            int ky = q / KW;
            int kx = q - ky * KW;
            int iy = iy0 + ky, ix = ix0 + kx;
            if ((unsigned)iy < (unsigned)H && (unsigned)ix < (unsigned)W)
              v = act[((size_t)(nimg * H + iy) * W + ix) * C + c];
          }
          Bs[r][kk + e] = v;
        }
      }
    }
    wait_async_lds();
    __syncthreads();

    // ---- fragments + WMMA ----
    v16h af[4];
#pragma unroll
    for (int mi = 0; mi < 4; ++mi) {
      const _Float16* ap = &As[wm * 64 + mi * 16 + (lane & 15)][(lane >> 4) * 16];
      union { v16h v; v8h h[2]; } u;
      u.h[0] = *(const v8h*)ap;
      u.h[1] = *(const v8h*)(ap + 8);
      af[mi] = u.v;
    }
#pragma unroll
    for (int ni = 0; ni < 2; ++ni) {
      const _Float16* bp = &Bs[wn * 32 + ni * 16 + (lane & 15)][(lane >> 4) * 16];
      v16h bf = *(const v16h*)bp;
#pragma unroll
      for (int mi = 0; mi < 4; ++mi)
        acc[mi][ni] = __builtin_amdgcn_wmma_f32_16x16x32_f16(
            false, af[mi], false, bf, (short)0, acc[mi][ni], false, false);
    }
    __syncthreads();
  }

  // ---- fused epilogue (NHWC): vector 16B stores per 8 channels ----
  const bool vecE = ((O & 7) == 0);
#pragma unroll
  for (int ni = 0; ni < 2; ++ni) {
    int p = p0 + wn * 32 + ni * 16 + (lane & 15);
    if (p >= P) continue;
    size_t base = (size_t)p * O;
#pragma unroll
    for (int mi = 0; mi < 4; ++mi) {
      int o0 = m0 + wm * 64 + mi * 16 + (lane >> 4) * 8;
      if (o0 >= O) continue;
      if (vecE) {
        v8h rh;
        if (resid) rh = *(const v8h*)(resid + base + o0);
        v8h hv;
#pragma unroll
        for (int e = 0; e < 8; ++e) {
          float v = acc[mi][ni][e] * bn_scale[o0 + e] + bn_bias[o0 + e];
          if (resid) v += h2f(rh[e]);
          if (relu && v < 0.0f) v = 0.0f;
          hv[e] = (_Float16)v;
          if (out_f) out_f[base + o0 + e] = v;
        }
        if (out_h) *(v8h*)(out_h + base + o0) = hv;
      } else {
#pragma unroll
        for (int e = 0; e < 8; ++e) {
          int o = o0 + e;
          if (o >= O) continue;
          float v = acc[mi][ni][e] * bn_scale[o] + bn_bias[o];
          if (resid) v += h2f(resid[base + o]);
          if (relu && v < 0.0f) v = 0.0f;
          if (out_h) out_h[base + o] = (_Float16)v;
          if (out_f) out_f[base + o] = v;
        }
      }
    }
  }
}

// ----------------------------- small kernels -------------------------------
// fp32 OIHW weight -> f16 O x (KH*KW*C)
__global__ void k_wcvt(const float* __restrict__ in, _Float16* __restrict__ out,
                       int O, int C, int KHW) {
  size_t i = (size_t)blockIdx.x * blockDim.x + threadIdx.x;
  size_t total = (size_t)O * C * KHW;
  if (i >= total) return;
  int q   = (int)(i % KHW);
  size_t t = i / KHW;
  int c   = (int)(t % C);
  int o   = (int)(t / C);
  out[((size_t)o * KHW + q) * C + c] = (_Float16)in[i];
}

// fp32 NCHW activation -> f16 NHWC
__global__ void k_x2nhwc(const float* __restrict__ in, _Float16* __restrict__ out,
                         int Nb, int C, int HW) {
  size_t i = (size_t)blockIdx.x * blockDim.x + threadIdx.x;
  size_t total = (size_t)Nb * C * HW;
  if (i >= total) return;
  int pix = (int)(i % HW);
  size_t t = i / HW;
  int c = (int)(t % C);
  int n = (int)(t / C);
  out[((size_t)n * HW + pix) * C + c] = (_Float16)in[i];
}

__global__ void k_bn_prep(const float* s, const float* b, const float* m, const float* v,
                          float eps, float* inv, float* bias, int C) {
  int i = blockIdx.x * blockDim.x + threadIdx.x;
  if (i < C) {
    float iv = rsqrtf(v[i] + eps) * s[i];
    inv[i]  = iv;
    bias[i] = b[i] - m[i] * iv;
  }
}

__global__ void k_scale_one(const float* b, float* inv, float* bias, int C) {
  int i = blockIdx.x * blockDim.x + threadIdx.x;
  if (i < C) { inv[i] = 1.0f; bias[i] = b[i]; }
}

// NHWC maxpool; thread index c-fastest (coalesced)
__global__ void k_maxpool(const _Float16* __restrict__ in, _Float16* __restrict__ out,
                          int Nb, int C, int H, int W, int k, int s, int pad,
                          int OH, int OW) {
  int i = blockIdx.x * blockDim.x + threadIdx.x;
  int total = Nb * C * OH * OW;
  if (i >= total) return;
  int c = i % C; int t = i / C;
  int ox = t % OW; t /= OW;
  int oy = t % OH; int n = t / OH;
  float best = -3.0e38f;
  for (int ky = 0; ky < k; ++ky) {
    int iy = oy * s - pad + ky;
    if ((unsigned)iy >= (unsigned)H) continue;
    for (int kx = 0; kx < k; ++kx) {
      int ix = ox * s - pad + kx;
      if ((unsigned)ix >= (unsigned)W) continue;
      float v = h2f(in[((size_t)(n * H + iy) * W + ix) * C + c]);
      best = v > best ? v : best;
    }
  }
  out[((size_t)(n * OH + oy) * OW + ox) * C + c] = (_Float16)best;
}

// NHWC global average pool: in (N,HW,C) -> out (N,C)
__global__ void k_avgpool(const _Float16* __restrict__ in, _Float16* __restrict__ out,
                          int Nb, int C, int HW) {
  int i = blockIdx.x * blockDim.x + threadIdx.x;
  if (i >= Nb * C) return;
  int c = i % C, n = i / C;
  float s = 0.0f;
  for (int j = 0; j < HW; ++j) s += h2f(in[((size_t)n * HW + j) * C + c]);
  out[i] = (_Float16)(s / (float)HW);
}

// head FC over NHWC pooled (32,7,7,128); k (reference NCHW flatten) = c*49+y*7+x
__global__ void k_head_fc(const _Float16* __restrict__ pooled, const float* __restrict__ w,
                          const float* __restrict__ b, float* __restrict__ out) {
  int o = blockIdx.x, n = blockIdx.y;
  __shared__ float red[256];
  float s = 0.0f;
  for (int k = threadIdx.x; k < 6272; k += 256) {
    int c = k / 49;
    int rem = k - c * 49;
    s += h2f(pooled[((size_t)n * 49 + rem) * 128 + c]) * w[(size_t)o * 6272 + k];
  }
  red[threadIdx.x] = s;
  __syncthreads();
  for (int st = 128; st > 0; st >>= 1) {
    if (threadIdx.x < st) red[threadIdx.x] += red[threadIdx.x + st];
    __syncthreads();
  }
  if (threadIdx.x == 0) out[n * 6 + o] = red[0] + b[o];
}

#define MIN_V (-4.765629f)
#define MAX_V (5.187813f)

__global__ void k_make_rot(const float* __restrict__ fc, float* __restrict__ R,
                           float* __restrict__ addv) {
  int n = threadIdx.x;
  if (n >= 32) return;
  const float* f = fc + n * 6;
  float ca = cosf(f[0]), sa = sinf(f[0]);
  float cb = cosf(f[1]), sb = sinf(f[1]);
  float cc = cosf(f[2]), sc = sinf(f[2]);
  float Rx[3][3] = {{1, 0, 0}, {0, ca, sa}, {0, -sa, ca}};
  float Ry[3][3] = {{cb, 0, -sb}, {0, 1, 0}, {sb, 0, cb}};
  float Rz[3][3] = {{cc, sc, 0}, {-sc, cc, 0}, {0, 0, 1}};
  float M[3][3], Rr[3][3];
  for (int i = 0; i < 3; ++i)
    for (int j = 0; j < 3; ++j) {
      float s = 0;
      for (int k = 0; k < 3; ++k) s += Rz[i][k] * Ry[k][j];
      M[i][j] = s;
    }
  for (int i = 0; i < 3; ++i)
    for (int j = 0; j < 3; ++j) {
      float s = 0;
      for (int k = 0; k < 3; ++k) s += M[i][k] * Rx[k][j];
      Rr[i][j] = s;
      R[n * 9 + i * 3 + j] = s;
    }
  float t[3] = {f[3] + MIN_V, f[4] + MIN_V, f[5] + MIN_V};
  for (int i = 0; i < 3; ++i) {
    float rv = Rr[i][0] * t[0] + Rr[i][1] * t[1] + Rr[i][2] * t[2];
    addv[n * 3 + i] = 255.0f * (rv - MIN_V) / (MAX_V - MIN_V);
  }
}

// x NCHW f32 -> sub NHWC f16: sub[n][pix][i] = R[n][i][:] . x[n][:][pix] + add[n][i]
__global__ void k_transform(const float* __restrict__ x, const float* __restrict__ R,
                            const float* __restrict__ addv, _Float16* __restrict__ sub,
                            int Nb, int HW) {
  int i = blockIdx.x * blockDim.x + threadIdx.x;
  if (i >= Nb * HW) return;
  int n = i / HW;
  int pix = i - n * HW;
  float x0 = x[((size_t)n * 3 + 0) * HW + pix];
  float x1 = x[((size_t)n * 3 + 1) * HW + pix];
  float x2 = x[((size_t)n * 3 + 2) * HW + pix];
  const float* Rn = R + n * 9;
  const float* an = addv + n * 3;
#pragma unroll
  for (int c = 0; c < 3; ++c)
    sub[((size_t)n * HW + pix) * 3 + c] =
        (_Float16)(Rn[c * 3] * x0 + Rn[c * 3 + 1] * x1 + Rn[c * 3 + 2] * x2 + an[c]);
}

// ------------------------------- host side ---------------------------------
extern "C" void kernel_launch(void* const* d_in, const int* in_sizes, int n_in,
                              void* d_out, int out_size, void* d_ws, size_t ws_size,
                              hipStream_t stream) {
  (void)in_sizes; (void)out_size; (void)ws_size; (void)n_in;

  char* ws = (char*)d_ws;
  size_t off = 0;
  auto alloc = [&](size_t bytes) -> void* {
    void* p = ws + off;
    off += (bytes + 255) & ~(size_t)255;
    return p;
  };
  auto cdiv = [](int a, int b) { return (a + b - 1) / b; };

  // activation arenas (largest tensor: 32*112*112*64 halfs = 51.4MB)
  const size_t SZh = (size_t)32 * 112 * 112 * 64;
  _Float16* arena = (_Float16*)alloc(5 * SZh * sizeof(_Float16));
  _Float16* A0 = arena;
  _Float16* A1 = arena + SZh;
  _Float16* A2 = arena + 2 * SZh;  // A2..A3 contiguous (head conv1 out spans both)
  _Float16* A3 = arena + 3 * SZh;
  _Float16* A4 = arena + 4 * SZh;

  // weight convert+reorder: OIHW f32 -> O x (KHW*C) f16
  auto cvt = [&](int idx, int O, int C, int KHW) -> _Float16* {
    size_t n = (size_t)O * C * KHW;
    _Float16* p = (_Float16*)alloc(n * sizeof(_Float16));
    k_wcvt<<<dim3((unsigned)((n + 255) / 256)), 256, 0, stream>>>(
        (const float*)d_in[idx], p, O, C, KHW);
    return p;
  };
  struct BnPtr { const float* inv; const float* bias; };
  auto bnp = [&](int idx, int C, float eps) -> BnPtr {
    float* inv  = (float*)alloc((size_t)C * 4);
    float* bias = (float*)alloc((size_t)C * 4);
    k_bn_prep<<<dim3(cdiv(C, 256)), 256, 0, stream>>>(
        (const float*)d_in[idx], (const float*)d_in[idx + 1],
        (const float*)d_in[idx + 2], (const float*)d_in[idx + 3], eps, inv, bias, C);
    return {inv, bias};
  };
  auto conv = [&](const _Float16* w, const _Float16* in, BnPtr bn,
                  const _Float16* resid, _Float16* outh, float* outf,
                  int C, int H, int O, int K, int stride, int pad, int relu) -> int {
    int OH = (H + 2 * pad - K) / stride + 1;
    int P  = 32 * OH * OH;
    dim3 g(cdiv(P, BN), cdiv(O, BM));
    k_conv_wmma<<<g, 256, 0, stream>>>(w, in, bn.inv, bn.bias, resid, outh, outf,
                                       32, C, H, H, O, K, K, stride, pad, OH, OH, relu);
    return OH;
  };

  // input x -> NHWC f16 (later overwritten with `sub`)
  const size_t xN = (size_t)32 * 3 * 224 * 224;
  _Float16* xh = (_Float16*)alloc(xN * sizeof(_Float16));
  k_x2nhwc<<<dim3((unsigned)((xN + 255) / 256)), 256, 0, stream>>>(
      (const float*)d_in[0], xh, 32, 3, 224 * 224);

  // ------------------------------ pose head --------------------------------
  _Float16* wc1 = cvt(1, 128, 3, 25);
  BnPtr hb1 = bnp(2, 128, 1e-3f);
  _Float16* wc2 = cvt(6, 128, 128, 25);
  BnPtr hb2 = bnp(7, 128, 1e-3f);

  conv(wc1, xh, hb1, nullptr, A2, nullptr, 3, 224, 128, 5, 2, 0, 1);    // (110,110,128)
  conv(wc2, A2, hb2, nullptr, A0, nullptr, 128, 110, 128, 5, 2, 0, 1);  // (53,53,128)

  _Float16* pooled = (_Float16*)alloc((size_t)32 * 7 * 7 * 128 * sizeof(_Float16));
  {
    int total = 32 * 128 * 7 * 7;
    k_maxpool<<<dim3(cdiv(total, 256)), 256, 0, stream>>>(A0, pooled, 32, 128, 53, 53, 7, 7, 0, 7, 7);
  }
  float* fco = (float*)alloc(192 * 4);
  k_head_fc<<<dim3(6, 32), 256, 0, stream>>>(pooled, (const float*)d_in[11],
                                             (const float*)d_in[12], fco);
  float* Rm  = (float*)alloc(32 * 9 * 4);
  float* adv = (float*)alloc(32 * 3 * 4);
  k_make_rot<<<1, 32, 0, stream>>>(fco, Rm, adv);
  k_transform<<<dim3(cdiv(32 * 224 * 224, 256)), 256, 0, stream>>>(
      (const float*)d_in[0], Rm, adv, xh, 32, 224 * 224);

  // ------------------------------ ResNet-50 --------------------------------
  const float EPS = 1e-5f;
  _Float16* wrc1 = cvt(13, 64, 3, 49);
  BnPtr rb1 = bnp(14, 64, EPS);
  conv(wrc1, xh, rb1, nullptr, A0, nullptr, 3, 224, 64, 7, 2, 3, 1);  // (112,112,64)
  {
    int total = 32 * 64 * 56 * 56;
    k_maxpool<<<dim3(cdiv(total, 256)), 256, 0, stream>>>(A0, A1, 32, 64, 112, 112, 3, 2, 1, 56, 56);
  }

  _Float16* cur = A1;
  _Float16* nxt = A0;
  int sp = 56, in_ch = 64;
  int idx = 18;
  const int nblocks[4] = {3, 4, 6, 3};
  for (int s = 0; s < 4; ++s) {
    int width  = 64 << s;
    int out_ch = width * 4;
    for (int j = 0; j < nblocks[s]; ++j) {
      int stride = (s > 0 && j == 0) ? 2 : 1;
      int base = idx;
      _Float16* w1 = cvt(base + 0, width, in_ch, 1);
      BnPtr q1 = bnp(base + 1, width, EPS);
      _Float16* w2 = cvt(base + 5, width, width, 9);
      BnPtr q2 = bnp(base + 6, width, EPS);
      _Float16* w3 = cvt(base + 10, out_ch, width, 1);
      BnPtr q3 = bnp(base + 11, out_ch, EPS);
      idx += 15;
      _Float16* wd = nullptr; BnPtr qd{nullptr, nullptr};
      if (j == 0) {
        wd = cvt(base + 15, out_ch, in_ch, 1);
        qd = bnp(base + 16, out_ch, EPS);
        idx += 5;
      }
      conv(w1, cur, q1, nullptr, A2, nullptr, in_ch, sp, width, 1, 1, 0, 1);
      int osp = conv(w2, A2, q2, nullptr, A3, nullptr, width, sp, width, 3, stride, 1, 1);
      const _Float16* resid = cur;
      if (j == 0) {
        conv(wd, cur, qd, nullptr, A4, nullptr, in_ch, sp, out_ch, 1, stride, 0, 0);
        resid = A4;
      }
      conv(w3, A3, q3, resid, nxt, nullptr, width, osp, out_ch, 1, 1, 0, 1);
      _Float16* t = cur; cur = nxt; nxt = t;
      sp = osp;
      in_ch = out_ch;
    }
  }

  // global average pool (NHWC 32,7,7,2048 -> 32,2048) into A2
  k_avgpool<<<cdiv(32 * 2048, 256), 256, 0, stream>>>(cur, A2, 32, 2048, 49);

  // final FC as 1x1 conv: O=1000, K=2048, P=32, f32 straight into d_out
  _Float16* wfc = cvt(idx, 1000, 2048, 1);
  float* fsc = (float*)alloc(1000 * 4);
  float* fbi = (float*)alloc(1000 * 4);
  k_scale_one<<<dim3(cdiv(1000, 256)), 256, 0, stream>>>((const float*)d_in[idx + 1], fsc, fbi, 1000);
  {
    BnPtr fb{fsc, fbi};
    conv(wfc, A2, fb, nullptr, nullptr, (float*)d_out, 2048, 1, 1000, 1, 1, 0, 0);
  }
}
